// Net_63754494542044
// MI455X (gfx1250) — matrix operand order repaired and checked by reading, not compiled, not run on hardware
//
#include <hip/hip_runtime.h>

typedef __attribute__((ext_vector_type(16))) _Float16 v16h;
typedef __attribute__((ext_vector_type(8)))  float    v8f;

union FragH { v16h h; uint4 q[2]; unsigned int u[8]; };

__device__ __forceinline__ v8f wmma_f16(v16h a, v16h b, v8f c) {
  return __builtin_amdgcn_wmma_f32_16x16x32_f16(false, a, false, b, (short)0, c, false, false);
}

// ---------------- f32 -> f16 convert ----------------
__global__ void cvt_f32_f16(const float* __restrict__ s, _Float16* __restrict__ d, int n) {
  int i = blockIdx.x * blockDim.x + threadIdx.x;
  if (i < n) d[i] = (_Float16)s[i];
}

// ---------------- conv2 weight reorder: OIHW f32 -> [oc][tap][ic] f16; also zero pad buffer ----------------
__global__ void reorder_w2(const float* __restrict__ w, _Float16* __restrict__ wb,
                           _Float16* __restrict__ zpad) {
  int i = blockIdx.x * blockDim.x + threadIdx.x;
  if (i < 64) zpad[i] = (_Float16)0.f;
  if (i >= 64 * 9 * 64) return;
  int ic = i & 63;
  int t  = (i >> 6) % 9;
  int oc = i / 576;
  wb[i] = (_Float16)w[((oc * 64 + ic) * 3 + t / 3) * 3 + (t % 3)];
}

// ---------------- conv1: direct, f32 VALU, output NHWC f16 ----------------
// x: [256][3][128][128], w: [64][3][3][3], out feat1: [256][64][64][64] (NHWC)
__global__ void conv1_k(const float* __restrict__ x, const float* __restrict__ w,
                        const float* __restrict__ b, _Float16* __restrict__ f1) {
  long tid = (long)blockIdx.x * blockDim.x + threadIdx.x;
  int oc  = (int)(tid & 63);
  int p   = (int)((tid >> 6) & 4095);
  int img = (int)(tid >> 18);
  if (img >= 256) return;
  int oy = p >> 6, ox = p & 63;
  float acc = b[oc];
  #pragma unroll
  for (int c = 0; c < 3; ++c)
    #pragma unroll
    for (int dy = 0; dy < 3; ++dy) {
      int iy = 2 * oy + dy;
      if (iy >= 128) continue;
      #pragma unroll
      for (int dx = 0; dx < 3; ++dx) {
        int ix = 2 * ox + dx;
        if (ix >= 128) continue;
        acc += x[(((size_t)img * 3 + c) * 128 + iy) * 128 + ix] *
               w[((oc * 3 + c) * 3 + dy) * 3 + dx];
      }
    }
  acc = fmaxf(acc, 0.f);
  f1[(((size_t)img * 64 + oy) * 64 + ox) * 64 + oc] = (_Float16)acc;
}

// ---------------- conv2: implicit GEMM via WMMA (branchless OOB via zero-pad pointer) ----------------
// feat1 NHWC [256][64][64][64] f16, wB [64oc][9tap][64ic] f16, out feat2 NHWC [256][1024][64]
__global__ __launch_bounds__(128) void conv2_wmma(const _Float16* __restrict__ feat1,
                                                  const _Float16* __restrict__ wB,
                                                  const float* __restrict__ bias,
                                                  const _Float16* __restrict__ zpad,
                                                  _Float16* __restrict__ feat2) {
  int lane = threadIdx.x & 31, wave = threadIdx.x >> 5;
  int mtile = blockIdx.x;          // 0..63 (16 pixels each)
  int img   = blockIdx.y;          // 0..255
  int hi = lane >> 4;
  int p  = mtile * 16 + (lane & 15);   // pixel 0..1023
  int oy = p >> 5, ox = p & 31;
  int oc = wave * 16 + (lane & 15);    // 0..63
  v8f acc = {};
  const _Float16* fimg = feat1 + (size_t)img * 64 * 64 * 64;
  for (int t = 0; t < 9; ++t) {
    int dy = t / 3, dx = t % 3;
    int iy = 2 * oy + dy, ix = 2 * ox + dx;
    bool ok = (iy < 64) && (ix < 64);          // SAME pad: lo=0, hi=1
    const _Float16* arow = ok ? (fimg + ((size_t)iy * 64 + ix) * 64) : zpad;
    const _Float16* wrow = wB + ((size_t)oc * 9 + t) * 64;
    #pragma unroll
    for (int h = 0; h < 2; ++h) {
      int icb = h * 32;
      FragH af, bf;
      af.q[0] = *(const uint4*)(arow + icb + hi * 8);
      af.q[1] = *(const uint4*)(arow + icb + hi * 8 + 16);
      bf.q[0] = *(const uint4*)(wrow + icb + hi * 8);
      bf.q[1] = *(const uint4*)(wrow + icb + hi * 8 + 16);
      acc = wmma_f16(af.h, bf.h, acc);
    }
  }
  float bv = bias[oc];
  #pragma unroll
  for (int r = 0; r < 8; ++r) {
    float val = fmaxf(acc[r] + bv, 0.f);
    int m = mtile * 16 + r + hi * 8;
    feat2[((size_t)img * 1024 + m) * 64 + oc] = (_Float16)val;
  }
}

// ---------------- generic GEMM: out = act(A[M,K] * W[N,K]^T + bias) ----------------
// N = 256 fixed per launch geometry: block = 128 (4 waves), each wave does NTPW n-tiles,
// reusing one A fragment per K-chunk for NTPW WMMAs. grid = (M/16, 1).
template <int NTPW>
__global__ __launch_bounds__(128) void gemm_wmma(const _Float16* __restrict__ A, int ldA,
                                                 const _Float16* __restrict__ Wt, int ldW,
                                                 const float* __restrict__ bias,
                                                 _Float16* __restrict__ outH, int ldOH,
                                                 float* __restrict__ outF, int ldOF,
                                                 int K, int relu) {
  int lane = threadIdx.x & 31, wave = threadIdx.x >> 5;
  int mtile = blockIdx.x;
  int hi = lane >> 4;
  int mrow = mtile * 16 + (lane & 15);
  int nlo  = lane & 15;
  const _Float16* Arow = A + (size_t)mrow * ldA + hi * 8;
  const _Float16* Wrow[NTPW];
  #pragma unroll
  for (int nt = 0; nt < NTPW; ++nt) {
    int ncol = (wave * NTPW + nt) * 16 + nlo;
    Wrow[nt] = Wt + (size_t)ncol * ldW + hi * 8;
  }
  v8f acc[NTPW];
  #pragma unroll
  for (int nt = 0; nt < NTPW; ++nt) acc[nt] = (v8f){};
  #pragma unroll 2
  for (int kb = 0; kb < K; kb += 32) {
    FragH af;
    af.q[0] = *(const uint4*)(Arow + kb);
    af.q[1] = *(const uint4*)(Arow + kb + 16);
    #pragma unroll
    for (int nt = 0; nt < NTPW; ++nt) {
      FragH bf;
      bf.q[0] = *(const uint4*)(Wrow[nt] + kb);
      bf.q[1] = *(const uint4*)(Wrow[nt] + kb + 16);
      acc[nt] = wmma_f16(af.h, bf.h, acc[nt]);
    }
  }
  #pragma unroll
  for (int nt = 0; nt < NTPW; ++nt) {
    int ncol = (wave * NTPW + nt) * 16 + nlo;
    float bv = bias ? bias[ncol] : 0.f;
    #pragma unroll
    for (int r = 0; r < 8; ++r) {
      float val = acc[nt][r] + bv;
      if (relu) val = fmaxf(val, 0.f);
      int m = mtile * 16 + r + hi * 8;
      if (outH) outH[(size_t)m * ldOH + ncol] = (_Float16)val;
      if (outF) outF[(size_t)m * ldOF + ncol] = val;
    }
  }
}

// ---------------- ROI align: feat2 NHWC f16 -> pool [1536][1024] f16 ----------------
__global__ void roi_k(const _Float16* __restrict__ f2, const float* __restrict__ rois,
                      _Float16* __restrict__ pool) {
  int r = blockIdx.x;
  const float* ro = rois + (size_t)r * 5;
  int bt = (int)ro[0];
  float x1 = ro[1] * 0.25f, y1 = ro[2] * 0.25f, x2 = ro[3] * 0.25f, y2 = ro[4] * 0.25f;
  float bw = fmaxf(x2 - x1, 1.f) * 0.25f;
  float bh = fmaxf(y2 - y1, 1.f) * 0.25f;
  const _Float16* fb = f2 + (size_t)bt * 32 * 32 * 64;
  for (int idx = threadIdx.x; idx < 1024; idx += blockDim.x) {
    int c = idx >> 4, pt = idx & 15, py = pt >> 2, px = pt & 3;
    float sx = x1 + bw * (px + 0.5f);
    float sy = y1 + bh * (py + 0.5f);
    float x0f = fminf(fmaxf(floorf(sx), 0.f), 31.f);
    float y0f = fminf(fmaxf(floorf(sy), 0.f), 31.f);
    float lx = fminf(fmaxf(sx - x0f, 0.f), 1.f);
    float ly = fminf(fmaxf(sy - y0f, 0.f), 1.f);
    int xx0 = (int)x0f, yy0 = (int)y0f;
    int xx1 = xx0 + 1 > 31 ? 31 : xx0 + 1;
    int yy1 = yy0 + 1 > 31 ? 31 : yy0 + 1;
    float f00 = (float)fb[((size_t)yy0 * 32 + xx0) * 64 + c];
    float f01 = (float)fb[((size_t)yy0 * 32 + xx1) * 64 + c];
    float f10 = (float)fb[((size_t)yy1 * 32 + xx0) * 64 + c];
    float f11 = (float)fb[((size_t)yy1 * 32 + xx1) * 64 + c];
    float v = f00 * (1.f - ly) * (1.f - lx) + f01 * (1.f - ly) * lx +
              f10 * ly * (1.f - lx) + f11 * ly * lx;
    pool[(size_t)r * 1024 + c * 16 + py * 4 + px] = (_Float16)v;
  }
}

// ---------------- coord embedding fc0c (K=2) ----------------
__global__ void emb0_k(const float* __restrict__ coor, const float* __restrict__ w,
                       const float* __restrict__ b, _Float16* __restrict__ e) {
  int tid = blockIdx.x * blockDim.x + threadIdx.x;   // 1536*256
  int row = tid >> 8, d = tid & 255;
  float v = coor[row * 2] * w[d * 2] + coor[row * 2 + 1] * w[d * 2 + 1] + b[d];
  e[tid] = (_Float16)fmaxf(v, 0.f);
}

// ---------------- scatter objred (B,T,N,D) -> 4 state slots (B,N,D); init coor slots ----------------
__global__ void init_state_k(const _Float16* __restrict__ objred,
                             const float* __restrict__ coor,
                             _Float16* __restrict__ states, float* __restrict__ coors) {
  int row = blockIdx.x, d = threadIdx.x;     // row = (b*4+t)*6+n
  int n = row % 6, bt = row / 6, t = bt & 3, b = bt >> 2;
  states[((size_t)t * 384 + b * 6 + n) * 256 + d] = objred[(size_t)row * 256 + d];
  if (d < 2) coors[((size_t)t * 384 + b * 6 + n) * 2 + d] = coor[(size_t)row * 2 + d];
}

// ---------------- r vector: per (b,n), mean over t of (h/2 + w/2)/2 ----------------
__global__ void rvec_k(const float* __restrict__ rois, float* __restrict__ rv) {
  int tid = blockIdx.x * blockDim.x + threadIdx.x;
  if (tid >= 384) return;
  int b = tid / 6, n = tid % 6;
  float acc = 0.f;
  for (int t = 0; t < 4; ++t) {
    const float* ro = rois + (size_t)(((b * 4 + t) * 6) + n) * 5;
    acc += ((ro[4] - ro[2]) * 0.5f + (ro[3] - ro[1]) * 0.5f) * 0.5f;
  }
  rv[tid] = acc * 0.25f;
}

// ---------------- masked relation reduce: affA = self_d + cnt*(u + rb) + sum_j mask*v_j ----------------
__global__ void relsum_k(const float* __restrict__ sd, const float* __restrict__ u,
                         const float* __restrict__ v, const float* __restrict__ rb,
                         const float* __restrict__ rv, const float* __restrict__ coor,
                         _Float16* __restrict__ affA) {
  int row = blockIdx.x, d = threadIdx.x;     // row = b*6+i
  int b = row / 6, i = row % 6;
  float cx = coor[(size_t)row * 2], cy = coor[(size_t)row * 2 + 1];
  float ri = rv[row];
  float acc = 0.f, cnt = 0.f;
  #pragma unroll
  for (int j = 0; j < 6; ++j) {
    if (j == i) continue;
    int rj = b * 6 + j;
    float dx = cx - coor[(size_t)rj * 2];
    float dy = cy - coor[(size_t)rj * 2 + 1];
    float dist = sqrtf(dx * dx + dy * dy);
    if (dist <= ri + rv[rj]) { acc += v[(size_t)rj * 256 + d]; cnt += 1.f; }
  }
  float val = sd[(size_t)row * 256 + d] + cnt * (u[(size_t)row * 256 + d] + rb[d]) + acc;
  affA[(size_t)row * 256 + d] = (_Float16)val;
}

// ---------------- copy state slot into right half of concat buffer (ld 512) ----------------
__global__ void copy_half_k(const _Float16* __restrict__ s, _Float16* __restrict__ dst) {
  int row = blockIdx.x, d = threadIdx.x;
  dst[(size_t)row * 512 + 256 + d] = s[(size_t)row * 256 + d];
}

// ---------------- decoder: bbox = s @ dec_w^T + dec_b; also update coor slot ----------------
__global__ void dec_k(const _Float16* __restrict__ s, const float* __restrict__ dw,
                      const float* __restrict__ db, float* __restrict__ out,
                      float* __restrict__ coorNew, int roll) {
  int tid = blockIdx.x * blockDim.x + threadIdx.x;
  if (tid >= 1536) return;
  int row = tid >> 2, o = tid & 3;
  float acc = db[o];
  const _Float16* sr = s + (size_t)row * 256;
  const float* wr = dw + (size_t)o * 256;
  for (int d = 0; d < 256; ++d) acc += (float)sr[d] * wr[d];
  int b = row / 6, n = row % 6;
  out[(((size_t)b * 8 + roll) * 6 + n) * 4 + o] = acc;
  if (o >= 2) coorNew[(size_t)row * 2 + (o - 2)] = acc;
}

extern "C" void kernel_launch(void* const* d_in, const int* in_sizes, int n_in,
                              void* d_out, int out_size, void* d_ws, size_t ws_size,
                              hipStream_t stream) {
  (void)in_sizes; (void)n_in; (void)out_size; (void)ws_size;
  const float* x        = (const float*)d_in[0];
  const float* rois     = (const float*)d_in[1];
  const float* scf      = (const float*)d_in[2];
  const float* w_conv1  = (const float*)d_in[3];
  const float* b_conv1  = (const float*)d_in[4];
  const float* w_conv2  = (const float*)d_in[5];
  const float* b_conv2  = (const float*)d_in[6];
  const float* fc0_w    = (const float*)d_in[7];
  const float* fc0_b    = (const float*)d_in[8];
  const float* fc0c_w   = (const float*)d_in[9];
  const float* fc0c_b   = (const float*)d_in[10];
  const float* fc1c_w   = (const float*)d_in[11];
  const float* fc1c_b   = (const float*)d_in[12];
  const float* red_w    = (const float*)d_in[13];
  const float* red_b    = (const float*)d_in[14];
  const float* g_self_w = (const float*)d_in[15];
  const float* g_self_b = (const float*)d_in[16];
  const float* g_rel_w  = (const float*)d_in[17];
  const float* g_rel_b  = (const float*)d_in[18];
  const float* g_aff_w  = (const float*)d_in[19];
  const float* g_aff_b  = (const float*)d_in[20];
  const float* g_out_w  = (const float*)d_in[21];
  const float* g_out_b  = (const float*)d_in[22];
  const float* agg_w    = (const float*)d_in[23];
  const float* agg_b    = (const float*)d_in[24];
  const float* dec_w    = (const float*)d_in[25];
  const float* dec_b    = (const float*)d_in[26];
  const int NROLL = 8;   // reference num_rollouts (device scalar unreadable under capture)

  char* base = (char*)d_ws;
  size_t off = 0;
  auto alloc = [&](size_t bytes) -> void* {
    void* p = base + off;
    off += (bytes + 255) & ~(size_t)255;
    return p;
  };
  _Float16* feat1h = (_Float16*)alloc(67108864ULL * 2);   // 256x64x64x64 NHWC
  _Float16* feat2h = (_Float16*)alloc(16777216ULL * 2);   // 256x32x32x64 NHWC
  _Float16* poolh  = (_Float16*)alloc(1536ULL * 1024 * 2);
  _Float16* emb1   = (_Float16*)alloc(1536ULL * 256 * 2);
  _Float16* cat0   = (_Float16*)alloc(1536ULL * 512 * 2);
  _Float16* objred = (_Float16*)alloc(1536ULL * 256 * 2);
  _Float16* states = (_Float16*)alloc(12ULL * 384 * 256 * 2);  // ring: 4 init + 8 generated
  float*    coors  = (float*)alloc(12ULL * 768 * 4);
  float*    rvec   = (float*)alloc(384ULL * 4);
  float*    sdbuf  = (float*)alloc(384ULL * 256 * 4);
  float*    ubuf   = (float*)alloc(384ULL * 256 * 4);
  float*    vbuf   = (float*)alloc(384ULL * 256 * 4);
  _Float16* affA   = (_Float16*)alloc(384ULL * 256 * 2);
  _Float16* catk   = (_Float16*)alloc(384ULL * 512 * 2);
  _Float16* catagg = (_Float16*)alloc(384ULL * 1024 * 2);
  _Float16* w2f    = (_Float16*)alloc(36864ULL * 2);
  _Float16* zpad   = (_Float16*)alloc(64ULL * 2);
  _Float16* fc0f   = (_Float16*)alloc(262144ULL * 2);
  _Float16* fc1cf  = (_Float16*)alloc(65536ULL * 2);
  _Float16* redf   = (_Float16*)alloc(131072ULL * 2);
  _Float16* selff  = (_Float16*)alloc(262144ULL * 2);
  _Float16* relf   = (_Float16*)alloc(524288ULL * 2);
  _Float16* afff   = (_Float16*)alloc(262144ULL * 2);
  _Float16* outfw  = (_Float16*)alloc(524288ULL * 2);
  _Float16* aggf   = (_Float16*)alloc(262144ULL * 2);

  auto cvt = [&](const float* s, _Float16* d, int n) {
    cvt_f32_f16<<<(n + 255) / 256, 256, 0, stream>>>(s, d, n);
  };
  cvt(fc0_w, fc0f, 262144);
  cvt(fc1c_w, fc1cf, 65536);
  cvt(red_w, redf, 131072);
  cvt(g_self_w, selff, 262144);
  cvt(g_rel_w, relf, 524288);
  cvt(g_aff_w, afff, 262144);
  cvt(g_out_w, outfw, 524288);
  cvt(agg_w, aggf, 262144);
  reorder_w2<<<(36864 + 255) / 256, 256, 0, stream>>>(w_conv2, w2f, zpad);

  conv1_k<<<262144, 256, 0, stream>>>(x, w_conv1, b_conv1, feat1h);
  conv2_wmma<<<dim3(64, 256), 128, 0, stream>>>(feat1h, w2f, b_conv2, zpad, feat2h);
  roi_k<<<1536, 256, 0, stream>>>(feat2h, rois, poolh);
  emb0_k<<<1536, 256, 0, stream>>>(scf, fc0c_w, fc0c_b, emb1);

  // obj = relu(fc0(pool)) -> cat0[:, :256]
  gemm_wmma<4><<<96, 128, 0, stream>>>(poolh, 1024, fc0f, 1024, fc0_b,
                                       cat0, 512, nullptr, 0, 1024, 1);
  // emb = relu(fc1c(emb1)) -> cat0[:, 256:]
  gemm_wmma<4><<<96, 128, 0, stream>>>(emb1, 256, fc1cf, 256, fc1c_b,
                                       cat0 + 256, 512, nullptr, 0, 256, 1);
  // objred = relu(red(cat0))
  gemm_wmma<4><<<96, 128, 0, stream>>>(cat0, 512, redf, 512, red_b,
                                       objred, 256, nullptr, 0, 512, 1);
  init_state_k<<<1536, 256, 0, stream>>>(objred, scf, states, coors);
  rvec_k<<<2, 256, 0, stream>>>(rois, rvec);

  for (int roll = 0; roll < NROLL; ++roll) {
    for (int k = 0; k < 4; ++k) {
      _Float16* S = states + (size_t)(roll + k) * 384 * 256;
      const float* C = coors + (size_t)(roll + k) * 768;
      // self_d = s @ g_self_w[k]^T + b   (f32)
      gemm_wmma<4><<<24, 128, 0, stream>>>(S, 256, selff + k * 65536, 256,
                                           g_self_b + k * 256,
                                           nullptr, 0, sdbuf, 256, 256, 0);
      // u = s @ W1^T, v = s @ W2^T  where g_rel_w[k] = [W1 | W2]
      gemm_wmma<4><<<24, 128, 0, stream>>>(S, 256, relf + k * 131072, 512, nullptr,
                                           nullptr, 0, ubuf, 256, 256, 0);
      gemm_wmma<4><<<24, 128, 0, stream>>>(S, 256, relf + k * 131072 + 256, 512, nullptr,
                                           nullptr, 0, vbuf, 256, 256, 0);
      relsum_k<<<384, 256, 0, stream>>>(sdbuf, ubuf, vbuf, g_rel_b + k * 256, rvec, C, affA);
      // a = relu(aff(self_d + rel)) -> catk[:, :256]
      gemm_wmma<4><<<24, 128, 0, stream>>>(affA, 256, afff + k * 65536, 256,
                                           g_aff_b + k * 256,
                                           catk, 512, nullptr, 0, 256, 1);
      copy_half_k<<<384, 256, 0, stream>>>(S, catk);   // catk[:, 256:] = s
      // cs[k] = relu(out(cat(a,s))) -> catagg[:, k*256 : (k+1)*256]
      gemm_wmma<4><<<24, 128, 0, stream>>>(catk, 512, outfw + k * 131072, 512,
                                           g_out_b + k * 256,
                                           catagg + k * 256, 1024, nullptr, 0, 512, 1);
    }
    _Float16* Snew = states + (size_t)(4 + roll) * 384 * 256;
    gemm_wmma<4><<<24, 128, 0, stream>>>(catagg, 1024, aggf, 1024, agg_b,
                                         Snew, 256, nullptr, 0, 1024, 0);
    dec_k<<<6, 256, 0, stream>>>(Snew, dec_w, dec_b, (float*)d_out,
                                 coors + (size_t)(4 + roll) * 768, roll);
  }
}
